// Self_Attention_v2_43817256353889
// MI455X (gfx1250) — matrix operand order repaired and checked by reading, not compiled, not run on hardware
//
#include <hip/hip_runtime.h>

// ---------------------------------------------------------------------------
// Self-attention (B=4, C=256, N=64*64=4096, E=32) for MI455X (gfx1250).
// Flash-attention formulation (S never materialized), all matmuls via
// v_wmma_f32_16x16x32_f16. K/V key blocks staged into LDS once per workgroup
// (8 waves share) via CDNA5 async global->LDS copies, double-buffered.
// ---------------------------------------------------------------------------

typedef _Float16 h16;
typedef __attribute__((ext_vector_type(2)))  _Float16 h2x;
typedef __attribute__((ext_vector_type(16))) _Float16 v16h;
typedef __attribute__((ext_vector_type(8)))  float    v8f;

// Pointer types for the async global->LDS builtin (param types taken from the
// compiler diagnostic: 'int __vector(4) __device__ *' i.e. vector_size(16)
// int in addrspace(1), and the LDS side in addrspace(3)).
typedef int v4i_vs __attribute__((vector_size(16)));
typedef __attribute__((address_space(1))) v4i_vs* gv4p;
typedef __attribute__((address_space(3))) v4i_vs* lv4p;

#define BB 4
#define CC 256
#define NN 4096
#define EE 32

#if defined(__has_builtin)
#if __has_builtin(__builtin_amdgcn_global_load_async_to_lds_b128) && \
    __has_builtin(__builtin_amdgcn_s_wait_asynccnt)
#define USE_ASYNC_LDS 1
#else
#define USE_ASYNC_LDS 0
#endif
#else
#define USE_ASYNC_LDS 0
#endif

static __device__ __forceinline__ v8f wmma_f16(v16h a, v16h b, v8f c) {
  return __builtin_amdgcn_wmma_f32_16x16x32_f16(false, a, false, b, (short)0, c,
                                                false, false);
}

// A-fragment (16x32, f16): lane holds row M = r0 + (lane&15);
// element s: K = s + 16*(lane>=16).
// Requires contiguous f16 pairs along K (row stride rs elements).
// Templated on pointer type so __shared__ roots keep LDS address space.
template <typename PT>
static __device__ __forceinline__ v16h load_A(PT base, int rs, int r0, int c0) {
  int lane = threadIdx.x & 31;
  int ln = lane & 15, lh = lane >> 4;
  PT rowp = base + (size_t)(r0 + ln) * rs + c0 + lh * 16;
  v16h a;
#pragma unroll
  for (int r = 0; r < 4; ++r) {
    a[2 * r]     = rowp[2 * r];
    a[2 * r + 1] = rowp[2 * r + 1];
    a[8 + 2 * r]     = rowp[8 + 2 * r];
    a[8 + 2 * r + 1] = rowp[8 + 2 * r + 1];
  }
  return a;
}

// B-fragment (32x16, f16): lane holds column N = n0 + (lane&15);
// lanes 0-15 hold K=0..15, lanes 16-31 hold K=16..31 (pairs per VGPR).
// Memory layout: for fixed column, K contiguous (row stride rs elements).
template <typename PT>
static __device__ __forceinline__ v16h load_B(PT base, int rs, int n0, int c0) {
  int lane = threadIdx.x & 31;
  int ln = lane & 15, lh = lane >> 4;
  PT p = base + (size_t)(n0 + ln) * rs + c0 + lh * 16;
  v16h b;
#pragma unroll
  for (int r = 0; r < 16; ++r) b[r] = p[r];
  return b;
}

// --------------------------- pass 1a: weights f32->f16 ----------------------
__global__ void k_wcvt(const float* __restrict__ Wk, const float* __restrict__ Wq,
                       const float* __restrict__ Wv, const float* __restrict__ Wo,
                       h16* __restrict__ Wkh, h16* __restrict__ Wqh,
                       h16* __restrict__ Wvh, h16* __restrict__ Woh) {
  int i = blockIdx.x * blockDim.x + threadIdx.x;
  if (i < EE * CC) {
    Wkh[i] = (h16)Wk[i];
    Wqh[i] = (h16)Wq[i];
    Wvh[i] = (h16)Wv[i];
    Woh[i] = (h16)Wo[i];   // Wo is [C,E] = 8192 elements too
  }
}

// --------------------------- pass 1b: x -> xT (f16, [N,C]) ------------------
__global__ void k_xt(const float* __restrict__ x, h16* __restrict__ xT) {
  __shared__ float tile[32][33];
  int b = blockIdx.z;
  int n0 = blockIdx.x * 32;
  int c0 = blockIdx.y * 32;
  const float* xb = x + (size_t)b * CC * NN;
  h16* xTb = xT + (size_t)b * NN * CC;
  int tx = threadIdx.x, ty = threadIdx.y;  // 32 x 8
#pragma unroll
  for (int i = 0; i < 32; i += 8)
    tile[ty + i][tx] = xb[(size_t)(c0 + ty + i) * NN + n0 + tx];
  __syncthreads();
#pragma unroll
  for (int i = 0; i < 32; i += 8)
    xTb[(size_t)(n0 + ty + i) * CC + c0 + tx] = (h16)tile[tx][ty + i];
}

// --------------------------- pass 2: QKV projections (WMMA) -----------------
// Kt = (Wk x + bk)^T as [N,E] f16 ; Qt likewise ; V = (Wv x + bv) as [E,N] f16.
__global__ void k_qkv(const h16* __restrict__ xT,
                      const h16* __restrict__ Wkh, const float* __restrict__ bk,
                      const h16* __restrict__ Wqh, const float* __restrict__ bq,
                      const h16* __restrict__ Wvh, const float* __restrict__ bv,
                      h16* __restrict__ Kt, h16* __restrict__ Qt,
                      h16* __restrict__ Vm) {
  int wave = threadIdx.x >> 5;
  int job = blockIdx.x * 4 + wave;      // B * N/16 = 1024 jobs
  int b = job >> 8;
  int n0 = (job & 255) * 16;
  int lane = threadIdx.x & 31;
  int ln = lane & 15, lh = lane >> 4;
  const h16* xTb = xT + (size_t)b * NN * CC;

  v8f aK0 = {}, aK1 = {}, aQ0 = {}, aQ1 = {}, aV0 = {}, aV1 = {};
#pragma unroll 2
  for (int c0 = 0; c0 < CC; c0 += 32) {
    v16h Ax  = load_A(xTb, CC, n0, c0);   // [query, c]
    v16h Bx  = load_B(xTb, CC, n0, c0);   // [c, query]
    v16h Bk0 = load_B(Wkh, CC, 0,  c0);   // [c, e0..15]
    v16h Bk1 = load_B(Wkh, CC, 16, c0);
    v16h Bq0 = load_B(Wqh, CC, 0,  c0);
    v16h Bq1 = load_B(Wqh, CC, 16, c0);
    v16h Av0 = load_A(Wvh, CC, 0,  c0);   // [e0..15, c]
    v16h Av1 = load_A(Wvh, CC, 16, c0);
    aK0 = wmma_f16(Ax, Bk0, aK0);
    aK1 = wmma_f16(Ax, Bk1, aK1);
    aQ0 = wmma_f16(Ax, Bq0, aQ0);
    aQ1 = wmma_f16(Ax, Bq1, aQ1);
    aV0 = wmma_f16(Av0, Bx, aV0);
    aV1 = wmma_f16(Av1, Bx, aV1);
  }

  h16* Ktb = Kt + (size_t)b * NN * EE;
  h16* Qtb = Qt + (size_t)b * NN * EE;
  h16* Vb  = Vm + (size_t)b * EE * NN;
  float bk0 = bk[ln], bk1 = bk[16 + ln];
  float bq0 = bq[ln], bq1 = bq[16 + ln];
#pragma unroll
  for (int r = 0; r < 8; ++r) {             // C-layout: row = r + 8*lh, col = ln
    int q = n0 + r + lh * 8;
    Ktb[q * EE + ln]      = (h16)(aK0[r] + bk0);
    Ktb[q * EE + 16 + ln] = (h16)(aK1[r] + bk1);
    Qtb[q * EE + ln]      = (h16)(aQ0[r] + bq0);
    Qtb[q * EE + 16 + ln] = (h16)(aQ1[r] + bq1);
  }
#pragma unroll
  for (int r = 0; r < 8; ++r) {
    int e = r + lh * 8;
    Vb[(size_t)e * NN + n0 + ln]        = (h16)(aV0[r] + bv[e]);
    Vb[(size_t)(16 + e) * NN + n0 + ln] = (h16)(aV1[r] + bv[16 + e]);
  }
}

// --------------------------- pass 3: flash attention ------------------------
// 256 threads = 8 waves, each wave one 16-query tile (same batch). K/V key
// block (32 keys) staged in LDS once per workgroup, double-buffered, copied
// with async global->LDS (ASYNCcnt) so the copy overlaps softmax + WMMA.
__global__ void __launch_bounds__(256) k_attn(const h16* __restrict__ Kt,
                                              const h16* __restrict__ Qt,
                                              const h16* __restrict__ Vm,
                                              h16* __restrict__ oT) {
  __shared__ __attribute__((aligned(16))) h16 Kbuf[2][32 * EE];  // [key][e]
  __shared__ __attribute__((aligned(16))) h16 Vbuf[2][EE * 32];  // [e][key]

  int wave = threadIdx.x >> 5;
  int b = blockIdx.x >> 5;                    // grid = B * (N/128) = 128
  int qbase = (blockIdx.x & 31) * 128 + wave * 16;
  int lane = threadIdx.x & 31;
  int ln = lane & 15, lh = lane >> 4;
  const h16* Ktb = Kt + (size_t)b * NN * EE;
  const h16* Qtb = Qt + (size_t)b * NN * EE;
  const h16* Vb  = Vm + (size_t)b * EE * NN;

  // Copy roles (wave-uniform): threads 0..127 stage K (2KB contiguous),
  // threads 128..255 stage V (32 rows x 64B, row stride NN*2 bytes).
  int t = threadIdx.x;
  bool isK = t < 128;
  int vt = t & 127;
  size_t g_off = isK ? (size_t)vt * 8
                     : (size_t)(vt >> 2) * NN + (size_t)(vt & 3) * 8;
  int l_off = isK ? vt * 8 : (vt >> 2) * 32 + (vt & 3) * 8;

  auto issue_copy = [&](int kb, int buf) {
    const h16* g = isK ? (Ktb + (size_t)kb * EE + g_off) : (Vb + kb + g_off);
    h16* l = (isK ? &Kbuf[buf][0] : &Vbuf[buf][0]) + l_off;
#if USE_ASYNC_LDS
    __builtin_amdgcn_global_load_async_to_lds_b128(
        (gv4p)(uintptr_t)g, (lv4p)(uintptr_t)l, 0, 0);
#else
    uint4 v = *(const uint4*)g;
    *(uint4*)l = v;
#endif
  };

  issue_copy(0, 0);
#if USE_ASYNC_LDS
  __builtin_amdgcn_s_wait_asynccnt(0);
#endif
  __syncthreads();

  v16h Bq = load_B(Qtb, EE, qbase, 0);  // [e, query] fragment, reused all loop
  v8f O0 = {}, O1 = {};
  float m = -3.0e38f, l = 0.0f;

  for (int kb = 0; kb < NN; kb += 32) {
    int cur = (kb >> 5) & 1;
    if (kb + 32 < NN) issue_copy(kb + 32, cur ^ 1);   // overlaps compute below

    v16h Af0 = load_A(&Kbuf[cur][0], EE, 0, 0);   // [key 0..15,  e]
    v16h Af1 = load_A(&Kbuf[cur][0], EE, 16, 0);  // [key 16..31, e]
    v16h Av0 = load_A(&Vbuf[cur][0], 32, 0, 0);   // [e 0..15,  key]
    v16h Av1 = load_A(&Vbuf[cur][0], 32, 16, 0);  // [e 16..31, key]

    v8f S0 = {}, S1 = {};
    S0 = wmma_f16(Af0, Bq, S0);                   // S[key, query]
    S1 = wmma_f16(Af1, Bq, S1);

    // online softmax over keys (column of S tile = 16 in-lane + partner half)
    float bm = S0[0];
#pragma unroll
    for (int r = 1; r < 8; ++r) bm = fmaxf(bm, S0[r]);
#pragma unroll
    for (int r = 0; r < 8; ++r) bm = fmaxf(bm, S1[r]);
    bm = fmaxf(bm, __shfl_xor(bm, 16, 32));
    float mnew = fmaxf(m, bm);
    float scale = __expf(m - mnew);
    float p0[8], p1[8], ps = 0.0f;
#pragma unroll
    for (int r = 0; r < 8; ++r) {
      p0[r] = __expf(S0[r] - mnew); ps += p0[r];
      p1[r] = __expf(S1[r] - mnew); ps += p1[r];
    }
    ps += __shfl_xor(ps, 16, 32);
    l = l * scale + ps;
    m = mnew;
#pragma unroll
    for (int r = 0; r < 8; ++r) { O0[r] *= scale; O1[r] *= scale; }

    // P: C-layout -> B-fragment layout (K=key, N=query) via lane-16 exchange
    v16h Bp;
#pragma unroll
    for (int r = 0; r < 4; ++r) {              // K = 2r (+16 for hi lanes)
      float x0 = __shfl_xor(p1[2 * r], 16, 32);
      float x1 = __shfl_xor(p1[2 * r + 1], 16, 32);
      Bp[2 * r]     = (h16)(lh ? x0 : p0[2 * r]);
      Bp[2 * r + 1] = (h16)(lh ? x1 : p0[2 * r + 1]);
    }
#pragma unroll
    for (int r = 4; r < 8; ++r) {              // K = 2r (+16 for hi lanes)
      int q = 2 * (r - 4);
      float x0 = __shfl_xor(p0[q], 16, 32);
      float x1 = __shfl_xor(p0[q + 1], 16, 32);
      Bp[2 * r]     = (h16)(lh ? p1[q] : x0);
      Bp[2 * r + 1] = (h16)(lh ? p1[q + 1] : x1);
    }

    O0 = wmma_f16(Av0, Bp, O0);
    O1 = wmma_f16(Av1, Bp, O1);

#if USE_ASYNC_LDS
    __builtin_amdgcn_s_wait_asynccnt(0);   // next block's copy landed (ours)
#endif
    __syncthreads();                       // publish copies; reads of cur done
  }

  float inv = 1.0f / l;
  h16* oTb = oT + (size_t)b * NN * EE;
#pragma unroll
  for (int r = 0; r < 8; ++r) {
    int e = r + lh * 8;
    oTb[(size_t)(qbase + ln) * EE + e]      = (h16)(O0[r] * inv);
    oTb[(size_t)(qbase + ln) * EE + 16 + e] = (h16)(O1[r] * inv);
  }
}

// --------------------------- pass 4: output projection ----------------------
// o = Wo v + bo (K = E = 32, exactly one WMMA per 16x16 tile); y = 0.5 o + x.
__global__ void k_out(const h16* __restrict__ Woh, const float* __restrict__ bo,
                      const h16* __restrict__ oT, const float* __restrict__ x,
                      float* __restrict__ out) {
  int wave = threadIdx.x >> 5;
  int job = blockIdx.x * 4 + wave;      // B * 16 ctiles * 16 ngroups = 1024
  int b = job >> 8;
  int rem = job & 255;
  int c0 = (rem >> 4) * 16;
  int ng = rem & 15;
  int lane = threadIdx.x & 31;
  int ln = lane & 15, lh = lane >> 4;

  v16h Aw = load_A(Woh, EE, c0, 0);     // [c, e] fragment, reused for 16 tiles
  const h16* oTb = oT + (size_t)b * NN * EE;
  const float* xb = x + (size_t)b * CC * NN;
  float* y_out = out;                           // y  at [0, B*C*N)
  float* o_out = out + (size_t)BB * CC * NN;    // o  at [B*C*N, 2*B*C*N)

  float biasr[8];
#pragma unroll
  for (int r = 0; r < 8; ++r) biasr[r] = bo[c0 + r + lh * 8];

#pragma unroll 4
  for (int tt = 0; tt < 16; ++tt) {
    int n0 = ng * 256 + tt * 16;
    v16h Bo = load_B(oTb, EE, n0, 0);   // [e, query]
    v8f acc = {};
    acc = wmma_f16(Aw, Bo, acc);
#pragma unroll
    for (int r = 0; r < 8; ++r) {
      int c = c0 + r + lh * 8;
      int n = n0 + ln;
      size_t idx = (size_t)b * CC * NN + (size_t)c * NN + n;
      float ov = acc[r] + biasr[r];
      o_out[idx] = ov;
      y_out[idx] = 0.5f * ov + xb[(size_t)c * NN + n];
    }
  }
  if (blockIdx.x == 0 && threadIdx.x == 0)
    out[(size_t)2 * BB * CC * NN] = 0.5f;       // gamma
}

// --------------------------- launcher ---------------------------------------
extern "C" void kernel_launch(void* const* d_in, const int* in_sizes, int n_in,
                              void* d_out, int out_size, void* d_ws, size_t ws_size,
                              hipStream_t stream) {
  const float* x  = (const float*)d_in[0];
  const float* Wk = (const float*)d_in[1];
  const float* bk = (const float*)d_in[2];
  const float* Wq = (const float*)d_in[3];
  const float* bq = (const float*)d_in[4];
  const float* Wv = (const float*)d_in[5];
  const float* bv = (const float*)d_in[6];
  const float* Wo = (const float*)d_in[7];
  const float* bo = (const float*)d_in[8];
  float* out = (float*)d_out;

  // workspace layout (bytes): ~12.1 MB total
  char* ws = (char*)d_ws;
  h16* xT  = (h16*)(ws);                    // [B][N][C] f16 :  8,388,608
  h16* Kt  = (h16*)(ws +  8388608);         // [B][N][E] f16 :  1,048,576
  h16* Qt  = (h16*)(ws +  9437184);         // [B][N][E] f16 :  1,048,576
  h16* Vm  = (h16*)(ws + 10485760);         // [B][E][N] f16 :  1,048,576
  h16* oT  = (h16*)(ws + 11534336);         // [B][N][E] f16 :  1,048,576
  h16* Wkh = (h16*)(ws + 12582912);         // [E][C] f16 : 16384
  h16* Wqh = (h16*)(ws + 12599296);
  h16* Wvh = (h16*)(ws + 12615680);
  h16* Woh = (h16*)(ws + 12632064);         // [C][E] f16 : 16384

  k_wcvt<<<32, 256, 0, stream>>>(Wk, Wq, Wv, Wo, Wkh, Wqh, Wvh, Woh);
  k_xt<<<dim3(NN / 32, CC / 32, BB), dim3(32, 8), 0, stream>>>(x, xT);
  k_qkv<<<256, 128, 0, stream>>>(xT, Wkh, bk, Wqh, bq, Wvh, bv, Kt, Qt, Vm);
  k_attn<<<128, 256, 0, stream>>>(Kt, Qt, Vm, oT);
  k_out<<<256, 128, 0, stream>>>(Woh, bo, oT, x, out);
}
